// SpatialCrossAttention_53747220742147
// MI455X (gfx1250) — matrix operand order repaired and checked by reading, not compiled, not run on hardware
//
#include <hip/hip_runtime.h>
#include <cmath>

typedef __attribute__((ext_vector_type(16))) _Float16 v16h;
typedef __attribute__((ext_vector_type(8)))  _Float16 v8h;
typedef __attribute__((ext_vector_type(8)))  float    v8f;

#define CC      256
#define HEADS   8
#define HH      46
#define WW      80
#define CAMS    6
#define KPTS    16
#define RADIUS_F 0.12f

// ---------------------------------------------------------------- f32 -> f16
__global__ void sca_cvt_f16(const float* __restrict__ in, _Float16* __restrict__ out, int n) {
    int i = blockIdx.x * blockDim.x + threadIdx.x;
    if (i < n) out[i] = (_Float16)in[i];
}

// ------------------------------------------ [CAMS,CC,H,W] -> [CAMS,H*W,CC]
__global__ void sca_transpose_feats(const float* __restrict__ in, float* __restrict__ out) {
    const int HWp = HH * WW;
    const long long total = (long long)CAMS * CC * HWp;
    long long o = (long long)blockIdx.x * blockDim.x + threadIdx.x;
    if (o >= total) return;
    int c  = (int)(o % CC);
    long long t = o / CC;
    int hw  = (int)(t % HWp);
    int cam = (int)(t / HWp);
    out[o] = in[((long long)cam * CC + c) * HWp + hw];
}

// -------------------------------------------------------------- WMMA GEMM
// out[M x cols] = A[M x 256] * Bw^T (+bias, epilogue). Bw stored [cols x 256].
// One wave per 16x16 tile; K=256 in 8 steps of v_wmma_f32_16x16x32_f16.
// Out-of-range rows are clamped to row 0 for loads (unconditional b128 loads,
// no per-lane exec churn) and never stored.
// mode 0: v = acc + bias ; mode 1: v = tanh(acc + bias) * RADIUS
__global__ void sca_gemm_wmma(const _Float16* __restrict__ A,
                              const _Float16* __restrict__ Bw,
                              const float*    __restrict__ bias,
                              float*          __restrict__ out,
                              int M, int cols, int colTiles, int totalTiles, int mode) {
    int wave = blockIdx.x * (blockDim.x >> 5) + (threadIdx.x >> 5);
    if (wave >= totalTiles) return;
    int lane = threadIdx.x & 31;
    int half = lane >> 4;          // 0: lanes 0-15, 1: lanes 16-31
    int r    = lane & 15;
    int row0 = (wave / colTiles) * 16;
    int col0 = (wave % colTiles) * 16;

    int arow = row0 + r;
    int lrow = (arow < M) ? arow : 0;                 // clamp for safe loads
    const _Float16* Abase = A  + (long long)lrow * CC;
    const _Float16* Bbase = Bw + (long long)(col0 + r) * CC;

    v8f acc = {};
    #pragma unroll
    for (int k0 = 0; k0 < CC; k0 += 32) {
        // A (16x32 f16): lane half holds K = {8h..8h+7} and {16+8h..16+8h+7}
        v8h alo = *(const v8h*)(Abase + k0 + 8 * half);
        v8h ahi = *(const v8h*)(Abase + k0 + 16 + 8 * half);
        v16h a = __builtin_shufflevector(alo, ahi,
                     0, 1, 2, 3, 4, 5, 6, 7, 8, 9, 10, 11, 12, 13, 14, 15);
        // B (32x16 f16): lane half holds K = {8h..8h+7} and {16+8h..16+8h+7} of column col0+r
        v8h blo = *(const v8h*)(Bbase + k0 + 8 * half);
        v8h bhi = *(const v8h*)(Bbase + k0 + 16 + 8 * half);
        v16h b = __builtin_shufflevector(blo, bhi,
                     0, 1, 2, 3, 4, 5, 6, 7, 8, 9, 10, 11, 12, 13, 14, 15);
        acc = __builtin_amdgcn_wmma_f32_16x16x32_f16(
                  false, a, false, b, (short)0, acc, false, false);
    }
    // D layout: VGPR i -> row = row0 + i + 8*half, col = col0 + r
    float bv = bias[col0 + r];
    #pragma unroll
    for (int i = 0; i < 8; ++i) {
        int row = row0 + i + 8 * half;
        if (row < M) {
            float v = acc[i] + bv;
            if (mode == 1) v = tanhf(v) * RADIUS_F;
            out[(long long)row * cols + col0 + r] = v;
        }
    }
}

// --------------------------------------------- sampling + softmax + fusion
// Block = one query (256 thr = 8 waves). Wave h = head h, lane = channel d.
// With featT ([cam,y,x,C]) each bilinear corner is a 128B contiguous wave read.
__global__ void sca_sampler(const float* __restrict__ sOff,   // [N,8,16,2]
                            const float* __restrict__ sWl,    // [N,8,16]
                            const float* __restrict__ feats,  // [CAMS,CC,H,W]
                            const float* __restrict__ featT,  // [CAMS,H,W,CC] or null
                            const float* __restrict__ refPts, // [CAMS,N,4,2]
                            const unsigned char* __restrict__ mask, // [CAMS,N,4]
                            _Float16* __restrict__ fusedH,    // [N,CC]
                            int N, int useT) {
    int q    = blockIdx.x;
    int h    = threadIdx.x >> 5;
    int lane = threadIdx.x & 31;
    int c    = h * 32 + lane;

    const float* off = sOff + (long long)q * 256 + h * 32; // [16][2]
    const float* wl  = sWl  + (long long)q * 128 + h * 16; // [16]

    float acc = 0.0f;
    int   cnt = 0;
    for (int cam = 0; cam < CAMS; ++cam) {
        const unsigned char* mk = mask + ((long long)cam * N + q) * 4;
        unsigned char m0 = mk[0], m1 = mk[1], m2 = mk[2], m3 = mk[3];
        if (!(m0 | m1 | m2 | m3)) continue;   // fully masked camera
        ++cnt;

        // masked softmax over K=16 (uniform across the wave; computed per lane)
        float lg[KPTS], wv[KPTS];
        float mx = -1e30f;
        #pragma unroll
        for (int k = 0; k < KPTS; ++k) {
            unsigned char vis = mk[k >> 2];
            float l = vis ? wl[k] : -1e30f;
            lg[k] = l;
            mx = fmaxf(mx, l);
        }
        float s = 0.0f;
        #pragma unroll
        for (int k = 0; k < KPTS; ++k) {
            float e = (lg[k] > -1e29f) ? __expf(lg[k] - mx) : 0.0f;
            wv[k] = e; s += e;
        }
        float inv = 1.0f / fmaxf(s, 1e-6f);

        const float* rp = refPts + ((long long)cam * N + q) * 8; // [4][2]
        #pragma unroll
        for (int k = 0; k < KPTS; ++k) {
            float w = wv[k] * inv;
            if (w == 0.0f) continue;          // wave-uniform skip of masked points
            int pip = k >> 2;
            float lx = rp[pip * 2 + 0] + off[k * 2 + 0];
            float ly = rp[pip * 2 + 1] + off[k * 2 + 1];
            // grid = loc*2-1 ; x = (gx+1)*0.5*W - 0.5  ==  loc*W - 0.5
            float x = lx * WW - 0.5f;
            float y = ly * HH - 0.5f;
            float x0f = floorf(x), y0f = floorf(y);
            int   x0 = (int)x0f,  y0 = (int)y0f;
            float fx = x - x0f,   fy = y - y0f;
            float sample = 0.0f;
            #pragma unroll
            for (int corner = 0; corner < 4; ++corner) {
                int dx = corner & 1, dy = corner >> 1;
                int xi = x0 + dx, yi = y0 + dy;
                // branch-free zero-padding: clamp address, zero the weight
                bool ok = ((unsigned)xi < (unsigned)WW) & ((unsigned)yi < (unsigned)HH);
                int xc = ok ? xi : 0;
                int yc = ok ? yi : 0;
                float cw = (dx ? fx : 1.0f - fx) * (dy ? fy : 1.0f - fy);
                cw = ok ? cw : 0.0f;
                float v = useT
                    ? featT[(((long long)cam * HH + yc) * WW + xc) * CC + c]
                    : feats[(((long long)cam * CC + c) * HH + yc) * WW + xc];
                sample += cw * v;
            }
            acc += w * sample;
        }
    }
    acc /= (float)(cnt < 1 ? 1 : cnt);
    fusedH[(long long)q * CC + c] = (_Float16)acc;
}

// ---------------------------------------------------------------- launcher
extern "C" void kernel_launch(void* const* d_in, const int* in_sizes, int n_in,
                              void* d_out, int out_size, void* d_ws, size_t ws_size,
                              hipStream_t stream) {
    const float* query    = (const float*)d_in[0];
    const float* feats    = (const float*)d_in[1];
    const float* refPts   = (const float*)d_in[2];
    const unsigned char* mask = (const unsigned char*)d_in[3];
    const float* offset_w = (const float*)d_in[4];
    const float* offset_b = (const float*)d_in[5];
    const float* weight_w = (const float*)d_in[6];
    const float* weight_b = (const float*)d_in[7];
    const float* out_w    = (const float*)d_in[8];
    const float* out_b    = (const float*)d_in[9];
    float* outp = (float*)d_out;

    const int N = in_sizes[0] / CC; // 2500

    // carve workspace (256B aligned)
    size_t o = 0;
    auto carve = [&](size_t bytes) { size_t p = o; o += (bytes + 255) & ~(size_t)255; return p; };
    char* ws = (char*)d_ws;
    size_t pQh   = carve((size_t)N * CC * 2);
    size_t pOffW = carve((size_t)CC * CC * 2);
    size_t pWtW  = carve((size_t)(HEADS * KPTS) * CC * 2);
    size_t pOutW = carve((size_t)CC * CC * 2);
    size_t pOff  = carve((size_t)N * CC * 4);
    size_t pWl   = carve((size_t)N * (HEADS * KPTS) * 4);
    size_t pFus  = carve((size_t)N * CC * 2);
    size_t featBytes = (size_t)CAMS * HH * WW * CC * 4;
    size_t pFeatT = o;
    int useT = (o + featBytes <= ws_size) ? 1 : 0;

    _Float16* sQh    = (_Float16*)(ws + pQh);
    _Float16* sOffW  = (_Float16*)(ws + pOffW);
    _Float16* sWtW   = (_Float16*)(ws + pWtW);
    _Float16* sOutW  = (_Float16*)(ws + pOutW);
    float*    sOff   = (float*)(ws + pOff);
    float*    sWl    = (float*)(ws + pWl);
    _Float16* sFus   = (_Float16*)(ws + pFus);
    float*    sFeatT = (float*)(ws + pFeatT);

    // 1) f32 -> f16 conversions
    {
        int n;
        n = N * CC;            sca_cvt_f16<<<(n + 255) / 256, 256, 0, stream>>>(query,    sQh,   n);
        n = CC * CC;           sca_cvt_f16<<<(n + 255) / 256, 256, 0, stream>>>(offset_w, sOffW, n);
        n = HEADS * KPTS * CC; sca_cvt_f16<<<(n + 255) / 256, 256, 0, stream>>>(weight_w, sWtW,  n);
        n = CC * CC;           sca_cvt_f16<<<(n + 255) / 256, 256, 0, stream>>>(out_w,    sOutW, n);
    }

    const int rowTiles = (N + 15) / 16;

    // 2) offsets = tanh(q @ offset_w.T + b) * RADIUS   -> sOff [N,256]
    {
        int colTiles = CC / 16, total = rowTiles * colTiles;
        sca_gemm_wmma<<<(total + 7) / 8, 256, 0, stream>>>(
            sQh, sOffW, offset_b, sOff, N, CC, colTiles, total, 1);
    }
    // 3) logits = q @ weight_w.T + b                   -> sWl [N,128]
    {
        int cols = HEADS * KPTS, colTiles = cols / 16, total = rowTiles * colTiles;
        sca_gemm_wmma<<<(total + 7) / 8, 256, 0, stream>>>(
            sQh, sWtW, weight_b, sWl, N, cols, colTiles, total, 0);
    }
    // 4) NHWC transpose of image features (if workspace allows)
    if (useT) {
        long long total = (long long)CAMS * CC * HH * WW;
        sca_transpose_feats<<<(unsigned)((total + 255) / 256), 256, 0, stream>>>(feats, sFeatT);
    }
    // 5) sample + masked softmax + camera fusion      -> sFus [N,256] f16
    sca_sampler<<<N, 256, 0, stream>>>(sOff, sWl, feats, useT ? sFeatT : nullptr,
                                       refPts, mask, sFus, N, useT);
    // 6) out = fused @ out_w.T + out_b                -> d_out [N,256] f32
    {
        int colTiles = CC / 16, total = rowTiles * colTiles;
        sca_gemm_wmma<<<(total + 7) / 8, 256, 0, stream>>>(
            sFus, sOutW, out_b, outp, N, CC, colTiles, total, 0);
    }
}
